// MultiHeadAttention_12326556140333
// MI455X (gfx1250) — matrix operand order repaired and checked by reading, not compiled, not run on hardware
//
#include <hip/hip_runtime.h>

typedef __attribute__((ext_vector_type(16))) __bf16 v16bf;
typedef __attribute__((ext_vector_type(8)))  __bf16 v8bf;
typedef __attribute__((ext_vector_type(8)))  float  v8f;
typedef __attribute__((ext_vector_type(4)))  unsigned int u32x4;
typedef __attribute__((ext_vector_type(8)))  int i32x8;
typedef __attribute__((ext_vector_type(4)))  int i32x4;

__device__ __forceinline__ v8f wmma_bf16(v16bf a, v16bf b, v8f c) {
  // D = A(16x32 bf16) x B(32x16 bf16) + C(16x16 f32)
  return __builtin_amdgcn_wmma_f32_16x16x32_bf16(false, a, false, b, (short)0, c,
                                                 false, false);
}

__device__ __forceinline__ v16bf cat8(v8bf lo, v8bf hi) {
  return __builtin_shufflevector(lo, hi, 0, 1, 2, 3, 4, 5, 6, 7,
                                 8, 9, 10, 11, 12, 13, 14, 15);
}

// A fragment (16x32 bf16) from row-major tile (ld elems): row m = lane&15,
// kc runs {hi*16..+7} and {hi*16+8..+7} -> two contiguous 16B LDS loads.
__device__ __forceinline__ v16bf load_a_frag(const __bf16* t, int ld, int lane) {
  const int hi = (lane >> 4) & 1, m = lane & 15;
  const v8bf lo = *(const v8bf*)&t[m * ld + hi * 16];
  const v8bf up = *(const v8bf*)&t[m * ld + hi * 16 + 8];
  return cat8(lo, up);
}

// B fragment (32x16 bf16) from TRANSPOSED tile sBt[n][kc] (ld = 32 elems):
// element j -> kc = hi*16 + j, all 16 contiguous -> one 32B LDS load.
__device__ __forceinline__ v16bf load_bT_frag(const __bf16* t, int lane) {
  const int hi = (lane >> 4) & 1, n = lane & 15;
  return *(const v16bf*)&t[n * 32 + hi * 16];
}

// ---- Tensor Data Mover: 2D bf16 tile (tileW x tileH, row stride in elems)
// D# per CDNA5 ISA 8.3/8.4; clang-23 6-arg builtin
// (u32x4 g0, i32x8 g1, i32x4 g2, i32x4 g3, i32x8 extra, i32 cpol)
__device__ __forceinline__ void tdm_load_tile_2d(unsigned ldsAddr, const void* gptr,
                                                 int tileW, int tileH,
                                                 long strideElems, long dim0Elems) {
  const unsigned long long ga = (unsigned long long)gptr;
  u32x4 g0;
  g0[0] = 1u;                                                   // count=1, user
  g0[1] = ldsAddr;                                              // lds_addr
  g0[2] = (unsigned)ga;                                         // global_addr lo
  g0[3] = (unsigned)((ga >> 32) & 0x01FFFFFFull) | (2u << 30);  // addr hi, type=2
  const unsigned d0 = (unsigned)dim0Elems;
  const unsigned d1 = 1u << 20;  // generous dim1 (tile always in-bounds)
  const unsigned long long s0 = (unsigned long long)strideElems;
  i32x8 g1;
  g1[0] = 0x00010000;                              // data_size=1 (2B), mask=0
  g1[1] = (int)((d0 & 0xFFFFu) << 16);             // tensor_dim0[15:0]
  g1[2] = (int)(((d0 >> 16) & 0xFFFFu) | ((d1 & 0xFFFFu) << 16));
  g1[3] = (int)(((d1 >> 16) & 0xFFFFu) | (((unsigned)tileW & 0xFFFFu) << 16));
  g1[4] = (int)((unsigned)tileH & 0xFFFFu);        // tile_dim1; tile_dim2=0
  g1[5] = (int)(unsigned)(s0 & 0xFFFFFFFFull);     // tensor_dim0_stride lo
  g1[6] = (int)(unsigned)((s0 >> 32) & 0xFFFFull); // stride hi; dim1_stride=0
  g1[7] = 0;
  const i32x4 z4 = {0, 0, 0, 0};
  const i32x8 z8 = {0, 0, 0, 0, 0, 0, 0, 0};
  __builtin_amdgcn_tensor_load_to_lds(g0, g1, z4, z4, z8, 0);
}

__global__ void cvt_f32_to_bf16_kernel(const float* __restrict__ in,
                                       __bf16* __restrict__ out, int n) {
  const int i = blockIdx.x * blockDim.x + threadIdx.x;
  if (i < n) out[i] = (__bf16)in[i];
}

// fp32 [K,N] row-major -> bf16 [N,K] row-major (weight transpose), 32x32 tiles
__global__ __launch_bounds__(256) void cvt_transpose_kernel(
    const float* __restrict__ in, __bf16* __restrict__ out, int K, int N) {
  __shared__ float tile[32][33];
  const int kt = blockIdx.y * 32, nt = blockIdx.x * 32;
  const int tx = threadIdx.x & 31, ty = threadIdx.x >> 5;  // 32 x 8
#pragma unroll
  for (int i = 0; i < 4; ++i)
    tile[ty + i * 8][tx] = in[(size_t)(kt + ty + i * 8) * N + nt + tx];
  __syncthreads();
#pragma unroll
  for (int i = 0; i < 4; ++i)
    out[(size_t)(nt + ty + i * 8) * K + kt + tx] = (__bf16)tile[tx][ty + i * 8];
}

// C[M,N] = A[M,K](bf16 rm) x BT[N,K](bf16 rm, pre-transposed) (+bias)
// TDM double-buffered pipeline: wave 0 issues tile k+1 into the spare LDS
// buffer while all waves run WMMAs on tile k; one barrier per k-step.
// MODE 0: bf16 row-major out.  MODE 1: fp32 out + bias.
// MODE 2: bf16 out transposed per head -> Vt[((b*16+h)*64+d)*2048 + t]
template <int MODE>
__global__ __launch_bounds__(128) void gemm_bf16_kernel(
    const __bf16* __restrict__ A, const __bf16* __restrict__ BT,
    const float* __restrict__ bias, float* __restrict__ Cf,
    __bf16* __restrict__ Cb, int M, int N, int K) {
  __shared__ __align__(32) __bf16 sA[2][64 * 32];
  __shared__ __align__(32) __bf16 sBt[2][64 * 32];
  const int tid = threadIdx.x, lane = tid & 31, wave = tid >> 5;
  const int wm = wave >> 1, wn = wave & 1;
  const int mBase = blockIdx.y * 64, nBase = blockIdx.x * 64;
  const unsigned ldsA[2] = {(unsigned)(size_t)(void*)&sA[0][0],
                            (unsigned)(size_t)(void*)&sA[1][0]};
  const unsigned ldsB[2] = {(unsigned)(size_t)(void*)&sBt[0][0],
                            (unsigned)(size_t)(void*)&sBt[1][0]};
  if (wave == 0) {  // prologue: fetch tile 0 into buffer 0
    tdm_load_tile_2d(ldsA[0], &A[(size_t)mBase * K], 32, 64, K, K);
    tdm_load_tile_2d(ldsB[0], &BT[(size_t)nBase * K], 32, 64, K, K);
  }
  v8f acc[2][2] = {};
  int cur = 0;
  for (int k0 = 0; k0 < K; k0 += 32) {
    if (wave == 0) __builtin_amdgcn_s_wait_tensorcnt(0);  // tile k landed
    __syncthreads();  // also proves buf[cur^1] readers (step k-1) are done
    if (wave == 0 && k0 + 32 < K) {  // overlap: DMA tile k+1 during compute
      tdm_load_tile_2d(ldsA[cur ^ 1], &A[(size_t)mBase * K + k0 + 32], 32, 64, K, K);
      tdm_load_tile_2d(ldsB[cur ^ 1], &BT[(size_t)nBase * K + k0 + 32], 32, 64, K, K);
    }
    const __bf16* cA = sA[cur];
    const __bf16* cB = sBt[cur];
    v16bf a0 = load_a_frag(&cA[(wm * 32 + 0) * 32], 32, lane);
    v16bf a1 = load_a_frag(&cA[(wm * 32 + 16) * 32], 32, lane);
    v16bf b0 = load_bT_frag(&cB[(wn * 32 + 0) * 32], lane);
    v16bf b1 = load_bT_frag(&cB[(wn * 32 + 16) * 32], lane);
    acc[0][0] = wmma_bf16(a0, b0, acc[0][0]);
    acc[0][1] = wmma_bf16(a0, b1, acc[0][1]);
    acc[1][0] = wmma_bf16(a1, b0, acc[1][0]);
    acc[1][1] = wmma_bf16(a1, b1, acc[1][1]);
    cur ^= 1;
  }
  const int hi = lane >> 4, lm = lane & 15;
#pragma unroll
  for (int mt = 0; mt < 2; ++mt)
#pragma unroll
    for (int nt = 0; nt < 2; ++nt) {
      const int col = nBase + wn * 32 + nt * 16 + lm;
      float bv = 0.f;
      if constexpr (MODE == 1) bv = bias[col];
      if constexpr (MODE == 2) {
        // rows r -> consecutive tokens: pack 8 bf16, one 16B store
        const int row0 = mBase + wm * 32 + mt * 16 + 8 * hi;
        const int bb = row0 >> 11, t0 = row0 & 2047;
        const int hh = col >> 6, dd = col & 63;
        v8bf pk;
#pragma unroll
        for (int r = 0; r < 8; ++r) pk[r] = (__bf16)acc[mt][nt][r];
        *(v8bf*)&Cb[((size_t)((bb * 16 + hh) * 64 + dd)) * 2048 + t0] = pk;
      } else {
#pragma unroll
        for (int r = 0; r < 8; ++r) {
          const int row = mBase + wm * 32 + mt * 16 + r + 8 * hi;
          const float val = acc[mt][nt][r] + bv;
          if constexpr (MODE == 1) Cf[(size_t)row * N + col] = val;
          else                     Cb[(size_t)row * N + col] = (__bf16)val;
        }
      }
    }
}

// One 32-key block of flash attention for a 16-row query tile.
template <bool MASKED>
__device__ __forceinline__ void attn_kblock(
    const __bf16* __restrict__ sK, const __bf16* __restrict__ sVt,
    __bf16* __restrict__ sPw, const v16bf& qa0, const v16bf& qa1,
    int k0, int q0, int lane, float* mrow, float* lrow, v8f* cacc) {
  const int hi = lane >> 4, lm = lane & 15;
  v8f s[2];
#pragma unroll
  for (int g = 0; g < 2; ++g) {
    const __bf16* krow = &sK[(g * 16 + lm) * 64];
    v16bf kf0 = *(const v16bf*)&krow[hi * 16];       // d = 0..31
    v16bf kf1 = *(const v16bf*)&krow[32 + hi * 16];  // d = 32..63
    v8f a = {};
    a = wmma_bf16(qa0, kf0, a);
    a = wmma_bf16(qa1, kf1, a);
    s[g] = a;
  }
#pragma unroll
  for (int r = 0; r < 8; ++r) {
    float s0 = s[0][r] * 0.125f;
    float s1 = s[1][r] * 0.125f;
    if (MASKED) {
      const int q = q0 + r + 8 * hi;
      if (k0 + lm > q)      s0 = -__builtin_inff();
      if (k0 + 16 + lm > q) s1 = -__builtin_inff();
    }
    float mx = fmaxf(s0, s1);
#pragma unroll
    for (int off = 8; off >= 1; off >>= 1)
      mx = fmaxf(mx, __shfl_xor(mx, off, 32));  // reduce within 16-lane half
    const float nm    = fmaxf(mrow[r], mx);
    const float alpha = __expf(mrow[r] - nm);
    const float p0 = __expf(s0 - nm), p1 = __expf(s1 - nm);
    float ps = p0 + p1;
#pragma unroll
    for (int off = 8; off >= 1; off >>= 1) ps += __shfl_xor(ps, off, 32);
    lrow[r] = lrow[r] * alpha + ps;
    mrow[r] = nm;
    cacc[0][r] *= alpha; cacc[1][r] *= alpha;
    cacc[2][r] *= alpha; cacc[3][r] *= alpha;
    sPw[(r + 8 * hi) * 32 + lm]      = (__bf16)p0;
    sPw[(r + 8 * hi) * 32 + 16 + lm] = (__bf16)p1;
  }
  const __bf16* prow = &sPw[lm * 32];
  v16bf pa = cat8(*(const v8bf*)&prow[hi * 16], *(const v8bf*)&prow[hi * 16 + 8]);
#pragma unroll
  for (int t = 0; t < 4; ++t) {
    v16bf vf = *(const v16bf*)&sVt[(t * 16 + lm) * 32 + hi * 16];
    cacc[t] = wmma_bf16(pa, vf, cacc[t]);
  }
}

// Flash attention: 4 waves / block, 64 query rows, shared K/V staging; causal.
// K token-major [tok, D]; V pre-transposed per head: Vt[((b*16+h)*64+d)*T + t]
__global__ __launch_bounds__(128) void flash_attn_kernel(
    const __bf16* __restrict__ Q, const __bf16* __restrict__ Kt,
    const __bf16* __restrict__ Vt, __bf16* __restrict__ O) {
  constexpr int T = 2048, DH = 64, D = 1024, H = 16;
  const int qblk = blockIdx.x, h = blockIdx.y, b = blockIdx.z;
  const int tid = threadIdx.x, lane = tid & 31, wave = tid >> 5;
  const int hi = lane >> 4, lm = lane & 15;
  const int q0 = qblk * 64 + wave * 16;  // this wave's 16-row query tile
  const size_t tokBase = (size_t)b * T;
  const __bf16* Qp = Q + tokBase * D + h * DH;
  const __bf16* Vp = Vt + (size_t)(b * H + h) * DH * T;

  __shared__ __align__(32) __bf16 sK[32 * 64];    // [key][d]
  __shared__ __align__(32) __bf16 sVt[64 * 32];   // [d][key]
  __shared__ __align__(32) __bf16 sP[4][16 * 32]; // per-wave [q][key]

  const __bf16* qrow = &Qp[(size_t)(q0 + lm) * D];
  v16bf qa0 = cat8(*(const v8bf*)&qrow[hi * 16], *(const v8bf*)&qrow[hi * 16 + 8]);
  v16bf qa1 = cat8(*(const v8bf*)&qrow[32 + hi * 16], *(const v8bf*)&qrow[40 + hi * 16]);

  float mrow[8], lrow[8];
#pragma unroll
  for (int r = 0; r < 8; ++r) { mrow[r] = -3.0e38f; lrow[r] = 0.f; }
  v8f cacc[4] = {};

  const int myKbMax  = (q0 + 15) >> 5;            // single partially-masked block
  const int blkKbMax = (qblk * 64 + 63) >> 5;
  for (int kb = 0; kb <= blkKbMax; ++kb) {
    const int k0 = kb * 32;
    // cooperative staging: 128 threads x 2 chunks x 16B per tile
#pragma unroll
    for (int i = 0; i < 2; ++i) {
      const int idx = tid * 2 + i;                 // 0..255
      const int kr = idx >> 3, kc = (idx & 7) * 8; // sK: 32 rows x 8 chunks
      *(uint4*)&sK[kr * 64 + kc] =
          *(const uint4*)&Kt[(tokBase + k0 + kr) * D + h * DH + kc];
      const int vr = idx >> 2, vc = (idx & 3) * 8; // sVt: 64 rows x 4 chunks
      *(uint4*)&sVt[vr * 32 + vc] = *(const uint4*)&Vp[(size_t)vr * T + k0 + vc];
    }
    if (kb < blkKbMax) {
      __builtin_prefetch(&Kt[(tokBase + k0 + 32 + (tid >> 2)) * D + h * DH], 0, 0);
      __builtin_prefetch(&Vp[(size_t)(tid >> 1) * T + k0 + 32], 0, 0);
    }
    __syncthreads();
    if (kb < myKbMax)                              // wave-uniform branches
      attn_kblock<false>(sK, sVt, sP[wave], qa0, qa1, k0, q0, lane, mrow, lrow, cacc);
    else if (kb == myKbMax)
      attn_kblock<true>(sK, sVt, sP[wave], qa0, qa1, k0, q0, lane, mrow, lrow, cacc);
    __syncthreads();
  }

  __bf16* Op = O + tokBase * D + h * DH;
#pragma unroll
  for (int t = 0; t < 4; ++t)
#pragma unroll
    for (int r = 0; r < 8; ++r)
      Op[(size_t)(q0 + r + 8 * hi) * D + t * 16 + lm] =
          (__bf16)(cacc[t][r] / lrow[r]);
}

extern "C" void kernel_launch(void* const* d_in, const int* in_sizes, int n_in,
                              void* d_out, int out_size, void* d_ws, size_t ws_size,
                              hipStream_t stream) {
  (void)in_sizes; (void)n_in; (void)out_size; (void)ws_size;
  const float* x  = (const float*)d_in[0];
  const float* Wq = (const float*)d_in[1];
  const float* Wk = (const float*)d_in[2];
  const float* Wv = (const float*)d_in[3];
  const float* Wo = (const float*)d_in[4];
  const float* bo = (const float*)d_in[5];

  constexpr int Bb = 2, T = 2048, D = 1024;
  constexpr int NT = Bb * T;  // 4096 token rows
  constexpr size_t MB = 1024ull * 1024ull;

  char* ws = (char*)d_ws;
  __bf16* xb  = (__bf16*)(ws + 0 * MB);   // 8 MB
  __bf16* wqT = (__bf16*)(ws + 8 * MB);   // 2 MB each, stored [N,K]
  __bf16* wkT = (__bf16*)(ws + 10 * MB);
  __bf16* wvT = (__bf16*)(ws + 12 * MB);
  __bf16* woT = (__bf16*)(ws + 14 * MB);
  __bf16* Qb  = (__bf16*)(ws + 16 * MB);  // 8 MB each
  __bf16* Kb  = (__bf16*)(ws + 24 * MB);
  __bf16* Vtb = (__bf16*)(ws + 32 * MB);  // V transposed per head [B,H,DH,T]
  __bf16* Cxb = (__bf16*)(ws + 40 * MB);  // ctx, 8 MB (48 MB total)

  const int nX = NT * D;
  cvt_f32_to_bf16_kernel<<<(nX + 255) / 256, 256, 0, stream>>>(x, xb, nX);
  dim3 tg(D / 32, D / 32);
  cvt_transpose_kernel<<<tg, 256, 0, stream>>>(Wq, wqT, D, D);
  cvt_transpose_kernel<<<tg, 256, 0, stream>>>(Wk, wkT, D, D);
  cvt_transpose_kernel<<<tg, 256, 0, stream>>>(Wv, wvT, D, D);
  cvt_transpose_kernel<<<tg, 256, 0, stream>>>(Wo, woT, D, D);

  dim3 gg(D / 64, NT / 64);  // (16, 64)
  gemm_bf16_kernel<0><<<gg, 128, 0, stream>>>(xb, wqT, nullptr, nullptr, Qb, NT, D, D);
  gemm_bf16_kernel<0><<<gg, 128, 0, stream>>>(xb, wkT, nullptr, nullptr, Kb, NT, D, D);
  gemm_bf16_kernel<2><<<gg, 128, 0, stream>>>(xb, wvT, nullptr, nullptr, Vtb, NT, D, D);

  flash_attn_kernel<<<dim3(T / 64, 16, Bb), 128, 0, stream>>>(Qb, Kb, Vtb, Cxb);

  gemm_bf16_kernel<1><<<gg, 128, 0, stream>>>(Cxb, woT, bo, (float*)d_out, nullptr, NT, D, D);
}
